// Knowledge_Decomposition_82686710382684
// MI455X (gfx1250) — matrix operand order repaired and checked by reading, not compiled, not run on hardware
//
#include <hip/hip_runtime.h>
#include <stdint.h>

#define D_DIM 256
#define MT 32          // rows (b*l) per block
#define AK 264         // padded bf16 row stride for staged A in LDS
#define CS 260         // padded f32 row stride for C in LDS
#define THREADS 256    // 8 waves (wave32)

typedef __attribute__((ext_vector_type(2)))  unsigned int u32x2;
typedef __attribute__((ext_vector_type(4)))  unsigned int u32x4;
typedef __attribute__((ext_vector_type(4)))  float        f32x4;
typedef __attribute__((ext_vector_type(8)))  float        f32x8;
typedef __attribute__((ext_vector_type(16))) __bf16       bf16x16;

union V16U { struct { u32x4 lo; u32x4 hi; } q; bf16x16 v; };

__device__ __forceinline__ unsigned short f2bf(float f) {
  union { float f; unsigned u; } c; c.f = f;
  unsigned u = c.u;
  unsigned r = u + 0x7FFFu + ((u >> 16) & 1u);   // round-to-nearest-even
  return (unsigned short)(r >> 16);
}
__device__ __forceinline__ unsigned pack_bf(float a, float b) {
  return (unsigned)f2bf(a) | ((unsigned)f2bf(b) << 16);
}
__device__ __forceinline__ float sigmoidf(float x) {
  return 1.0f / (1.0f + __expf(-x));
}

// ---- Prepass: fp32 weights -> bf16 in workspace (Wg then Wp, [2][256][256] each)
__global__ void kd_convert_weights(const float* __restrict__ Wg,
                                   const float* __restrict__ Wp,
                                   unsigned short* __restrict__ dst, int n) {
  int i = blockIdx.x * blockDim.x + threadIdx.x;
  if (i < n) {
    dst[i]     = f2bf(Wg[i]);
    dst[n + i] = f2bf(Wp[i]);
  }
}

// ---- Main fused kernel: GEMM(bf16 WMMA, f32 acc) + bias + LayerNorm + attention gating
__global__ __launch_bounds__(THREADS)
void kd_main(const float* __restrict__ gfeat, const float* __restrict__ pfeat,
             const unsigned short* __restrict__ WgBf, const unsigned short* __restrict__ WpBf,
             const float* __restrict__ bg,  const float* __restrict__ gng, const float* __restrict__ gnb,
             const float* __restrict__ bp,  const float* __restrict__ png, const float* __restrict__ pnb,
             const float* __restrict__ wga, const float* __restrict__ bga,
             const float* __restrict__ wpa, const float* __restrict__ bpa,
             float* __restrict__ out, int BL)
{
  const int e = blockIdx.y;
  const int rowBase = blockIdx.x * MT;
  extern __shared__ char smem[];
  unsigned short* aLds = (unsigned short*)smem;                               // [2][MT][AK] bf16
  float* cLds = (float*)(smem + (size_t)2 * MT * AK * sizeof(unsigned short)); // [2][MT][CS] f32

  const int tid = threadIdx.x;

  // ---- Stage A fp32 -> bf16 into LDS. mat0 = pfeat (feeds g), mat1 = gfeat (feeds p)
  #pragma unroll
  for (int mat = 0; mat < 2; ++mat) {
    const float* src = (mat == 0) ? pfeat : gfeat;
    #pragma unroll
    for (int it = 0; it < (MT * D_DIM / 4) / THREADS; ++it) {
      int i   = it * THREADS + tid;     // float4 index
      int row = i >> 6;                 // 64 float4 per row
      int col = (i & 63) << 2;
      f32x4 v = *(const f32x4*)(src + (size_t)(rowBase + row) * D_DIM + col);
      u32x2 pk;
      pk.x = pack_bf(v.x, v.y);
      pk.y = pack_bf(v.z, v.w);
      *(u32x2*)(aLds + (mat * MT + row) * AK + col) = pk;
    }
  }
  __syncthreads();

  // ---- WMMA GEMM. wave -> 2 col tiles; tiles: [mat][rowTile][colTile]
  const int wave = tid >> 5, lane = tid & 31;
  const int h = lane >> 4, n16 = lane & 15;
  const unsigned short* Wb[2];
  Wb[0] = WgBf + (size_t)e * (D_DIM * D_DIM);
  Wb[1] = WpBf + (size_t)e * (D_DIM * D_DIM);

  f32x8 acc[2][2][2] = {};

  for (int kk = 0; kk < D_DIM; kk += 32) {
    // A frags (ISA 16x32 bf16 A layout): lane m=n16, half h holds K in
    // [kk+8h, kk+8h+8) and [kk+16+8h, kk+16+8h+8)
    V16U af[2][2];
    #pragma unroll
    for (int mat = 0; mat < 2; ++mat)
      #pragma unroll
      for (int rt = 0; rt < 2; ++rt) {
        const unsigned short* base = aLds + (mat * MT + rt * 16 + n16) * AK + kk + 8 * h;
        af[mat][rt].q.lo = *(const u32x4*)(base);
        af[mat][rt].q.hi = *(const u32x4*)(base + 16);
      }
    // B frags (32x16 bf16 B layout): lane col n16, half h holds K in [kk+8h, kk+8h+8) and [kk+16+8h, kk+16+8h+8)
    // = W[col][kk+8h ..] and W[col][kk+16+8h ..] (W row-major == B^T)
    V16U bfr[2][2];
    #pragma unroll
    for (int mat = 0; mat < 2; ++mat)
      #pragma unroll
      for (int ci = 0; ci < 2; ++ci) {
        int colg = (wave * 2 + ci) * 16 + n16;
        const unsigned short* wr = Wb[mat] + (size_t)colg * D_DIM + kk + 8 * h;
        bfr[mat][ci].q.lo = *(const u32x4*)(wr);
        bfr[mat][ci].q.hi = *(const u32x4*)(wr + 16);
      }
    #pragma unroll
    for (int mat = 0; mat < 2; ++mat)
      #pragma unroll
      for (int rt = 0; rt < 2; ++rt)
        #pragma unroll
        for (int ci = 0; ci < 2; ++ci)
          acc[mat][rt][ci] = __builtin_amdgcn_wmma_f32_16x16x32_bf16(
              false, af[mat][rt].v, false, bfr[mat][ci].v,
              (short)0, acc[mat][rt][ci], false, false);
  }

  // ---- C (+bias) -> LDS (ISA f32 C layout: VGPR v = row 8h+v, lane col n16)
  #pragma unroll
  for (int mat = 0; mat < 2; ++mat) {
    const float* bvec = (mat == 0) ? bg : bp;
    #pragma unroll
    for (int ci = 0; ci < 2; ++ci) {
      int colg = (wave * 2 + ci) * 16 + n16;
      float bias = bvec[e * D_DIM + colg];
      #pragma unroll
      for (int rt = 0; rt < 2; ++rt)
        #pragma unroll
        for (int v = 0; v < 8; ++v) {
          int row = rt * 16 + 8 * h + v;
          cLds[(mat * MT + row) * CS + colg] = acc[mat][rt][ci][v] + bias;
        }
    }
  }
  __syncthreads();

  // ---- Epilogue: 8 threads per row (same wave), 32 cols each
  const int r = tid >> 3, seg = tid & 7, c0 = seg * 32;
  const float* gRow = cLds + (0 * MT + r) * CS;
  const float* pRow = cLds + (1 * MT + r) * CS;

  float gv_[32], pv_[32];
  float sg = 0.f, sg2 = 0.f, sp = 0.f, sp2 = 0.f;
  #pragma unroll
  for (int jj = 0; jj < 8; ++jj) {
    f32x4 gx = *(const f32x4*)(gRow + c0 + jj * 4);
    f32x4 px = *(const f32x4*)(pRow + c0 + jj * 4);
    #pragma unroll
    for (int k = 0; k < 4; ++k) {
      float x = gx[k], y = px[k];
      gv_[jj * 4 + k] = x; pv_[jj * 4 + k] = y;
      sg += x; sg2 += x * x; sp += y; sp2 += y * y;
    }
  }
  #pragma unroll
  for (int m = 1; m < 8; m <<= 1) {
    sg += __shfl_xor(sg, m);  sg2 += __shfl_xor(sg2, m);
    sp += __shfl_xor(sp, m);  sp2 += __shfl_xor(sp2, m);
  }
  const float inv = 1.0f / 256.0f;
  float gm = sg * inv, gva = sg2 * inv - gm * gm;
  float pm = sp * inv, pva = sp2 * inv - pm * pm;
  float grs = rsqrtf(gva + 1e-5f), prs = rsqrtf(pva + 1e-5f);

  const float* gngE = gng + e * D_DIM; const float* gnbE = gnb + e * D_DIM;
  const float* pngE = png + e * D_DIM; const float* pnbE = pnb + e * D_DIM;
  const float* wgaE = wga + e * D_DIM; const float* wpaE = wpa + e * D_DIM;

  float dga = 0.f, dpa = 0.f;
  #pragma unroll
  for (int jj = 0; jj < 8; ++jj) {
    int c = c0 + jj * 4;
    f32x4 ga_ = *(const f32x4*)(gngE + c); f32x4 gb_ = *(const f32x4*)(gnbE + c);
    f32x4 pa_ = *(const f32x4*)(pngE + c); f32x4 pb_ = *(const f32x4*)(pnbE + c);
    f32x4 wa_ = *(const f32x4*)(wgaE + c); f32x4 wp_ = *(const f32x4*)(wpaE + c);
    #pragma unroll
    for (int k = 0; k < 4; ++k) {
      int j = jj * 4 + k;
      float gn = (gv_[j] - gm) * grs * ga_[k] + gb_[k];
      float pn = (pv_[j] - pm) * prs * pa_[k] + pb_[k];
      gv_[j] = gn; pv_[j] = pn;
      dga += pn * wa_[k];   // dot(p_n, wga) -> gates g
      dpa += gn * wp_[k];   // dot(g_n, wpa) -> gates p
    }
  }
  #pragma unroll
  for (int m = 1; m < 8; m <<= 1) { dga += __shfl_xor(dga, m); dpa += __shfl_xor(dpa, m); }

  const float cga = bga[e], cpa = bpa[e];
  float* orow = out + ((size_t)e * BL + rowBase + r) * D_DIM;
  #pragma unroll
  for (int jj = 0; jj < 8; ++jj) {
    f32x4 o;
    #pragma unroll
    for (int k = 0; k < 4; ++k) {
      int j = jj * 4 + k;
      float gatt = sigmoidf(gv_[j] * dga + cga);
      float patt = sigmoidf(pv_[j] * dpa + cpa);
      o[k] = pv_[j] * patt + gv_[j] * gatt;
    }
    *(f32x4*)(orow + c0 + jj * 4) = o;
  }
}

extern "C" void kernel_launch(void* const* d_in, const int* in_sizes, int n_in,
                              void* d_out, int out_size, void* d_ws, size_t ws_size,
                              hipStream_t stream) {
  const float* gfeat = (const float*)d_in[0];
  const float* pfeat = (const float*)d_in[1];
  const float* Wg    = (const float*)d_in[2];
  const float* bg    = (const float*)d_in[3];
  const float* gng   = (const float*)d_in[4];
  const float* gnb   = (const float*)d_in[5];
  const float* Wp    = (const float*)d_in[6];
  const float* bp    = (const float*)d_in[7];
  const float* png   = (const float*)d_in[8];
  const float* pnb   = (const float*)d_in[9];
  const float* wga   = (const float*)d_in[10];
  const float* bga   = (const float*)d_in[11];
  const float* wpa   = (const float*)d_in[12];
  const float* bpa   = (const float*)d_in[13];
  float* out = (float*)d_out;

  const int nW = in_sizes[2];                      // 2*256*256 = 131072
  unsigned short* WgBf = (unsigned short*)d_ws;    // bf16 weights in workspace
  unsigned short* WpBf = WgBf + nW;

  kd_convert_weights<<<dim3((nW + 255) / 256), dim3(256), 0, stream>>>(Wg, Wp, WgBf, nW);

  const int BL = in_sizes[0] / D_DIM;              // 65536 rows
  const size_t smem = (size_t)2 * MT * AK * sizeof(unsigned short)
                    + (size_t)2 * MT * CS * sizeof(float);
  kd_main<<<dim3(BL / MT, 2), dim3(THREADS), smem, stream>>>(
      gfeat, pfeat, WgBf, WpBf, bg, gng, gnb, bp, png, pnb,
      wga, bga, wpa, bpa, out, BL);
}
